// RoPEAttention_39058432590045
// MI455X (gfx1250) — matrix operand order repaired and checked by reading, not compiled, not run on hardware
//
#include <hip/hip_runtime.h>
#include <hip/hip_bf16.h>

#define BQ 16
#define NQ 1025
#define DIMQ 768
#define HQ 12
#define HDQ 64
#define NP 1056   // padded sequence length: 33 * 32
#define NTOK (BQ * NQ)  // 16400

typedef __attribute__((ext_vector_type(16))) __bf16 v16bf;
typedef __attribute__((ext_vector_type(8)))  float  v8f;

__device__ __forceinline__ __bf16 f2bf(float f) {
    union { float f; unsigned u; } a; a.f = f;
    unsigned r = a.u + 0x7FFFu + ((a.u >> 16) & 1u);
    union { unsigned short s; __bf16 b; } o; o.s = (unsigned short)(r >> 16);
    return o.b;
}

__device__ __forceinline__ v16bf pack_frag(uint4 a, uint4 b) {
    union { uint4 u[2]; v16bf v; } t; t.u[0] = a; t.u[1] = b; return t.v;
}

// A operand: 16 rows x 32 K of a row-major bf16 matrix (stride ld elements).
// ISA layout: lanes 0-15 row=lane, K in {0..7, 16..23}; lanes 16-31 K in {8..15, 24..31}.
__device__ __forceinline__ v16bf load_a_frag(const __bf16* base, int ld) {
    int lane = threadIdx.x & 31;
    int row = lane & 15, g = lane >> 4;
    const __bf16* p = base + (size_t)row * ld + g * 8;
    uint4 c0 = *reinterpret_cast<const uint4*>(p);
    uint4 c1 = *reinterpret_cast<const uint4*>(p + 16);
    return pack_frag(c0, c1);
}

// B operand: 32 K x 16 cols where columns are rows of a row-major matrix.
// Lane holds column = lane&15, K in {0..7, 16..23} (lanes 0-15) or {8..15, 24..31} (lanes 16-31).
__device__ __forceinline__ v16bf load_b_frag(const __bf16* base, int ld) {
    int lane = threadIdx.x & 31;
    int col = lane & 15, g = lane >> 4;
    const __bf16* p = base + (size_t)col * ld + g * 8;
    uint4 c0 = *reinterpret_cast<const uint4*>(p);
    uint4 c1 = *reinterpret_cast<const uint4*>(p + 16);
    return pack_frag(c0, c1);
}

__device__ __forceinline__ v8f wmma_bf16(v16bf a, v16bf b, v8f c) {
    return __builtin_amdgcn_wmma_f32_16x16x32_bf16(false, a, false, b, (short)0, c,
                                                   false, false);
}

__device__ __forceinline__ float hmax16(float v) {
    v = fmaxf(v, __shfl_xor(v, 1, 32));
    v = fmaxf(v, __shfl_xor(v, 2, 32));
    v = fmaxf(v, __shfl_xor(v, 4, 32));
    v = fmaxf(v, __shfl_xor(v, 8, 32));
    return v;
}
__device__ __forceinline__ float hsum16(float v) {
    v += __shfl_xor(v, 1, 32);
    v += __shfl_xor(v, 2, 32);
    v += __shfl_xor(v, 4, 32);
    v += __shfl_xor(v, 8, 32);
    return v;
}

// ---------------- stage 0: convert fp32 -> bf16, zero pad buffers --------------
__global__ void cvt_bf16_kernel(const float* __restrict__ in, __bf16* __restrict__ out, int n) {
    for (int i = blockIdx.x * blockDim.x + threadIdx.x; i < n; i += gridDim.x * blockDim.x)
        out[i] = f2bf(in[i]);
}
__global__ void zero_kernel(unsigned* __restrict__ p, long long n) {
    for (long long i = blockIdx.x * (long long)blockDim.x + threadIdx.x; i < n;
         i += (long long)gridDim.x * blockDim.x)
        p[i] = 0u;
}

// ---------------- stage 1: QKV GEMM (64x32 per wave) + RoPE --------------------
// grid = MT(257) * ET(72); each wave: 4x2 WMMA tiles, K-loop over 768.
__global__ __launch_bounds__(32) void qkv_rope_kernel(
    const __bf16* __restrict__ xbf, const __bf16* __restrict__ wbf,
    const float* __restrict__ fcos, const float* __restrict__ fsin,
    __bf16* __restrict__ qbuf, __bf16* __restrict__ kbuf, __bf16* __restrict__ vT)
{
    const int ET = (3 * DIMQ) / 32;  // 72
    int blk = blockIdx.x;
    int mt = blk / ET, et = blk % ET;
    int m0 = mt * 64, e0 = et * 32;
    const __bf16* abase = xbf + (size_t)m0 * DIMQ;
    const __bf16* bbase = wbf + (size_t)e0 * DIMQ;

    const v8f vzero = {0, 0, 0, 0, 0, 0, 0, 0};
    v8f acc[4][2];
#pragma unroll
    for (int i = 0; i < 4; i++)
#pragma unroll
        for (int j = 0; j < 2; j++) acc[i][j] = vzero;

    for (int k = 0; k < DIMQ; k += 32) {
        v16bf af[4], bfr[2];
#pragma unroll
        for (int i = 0; i < 4; i++)
            af[i] = load_a_frag(abase + (size_t)(i * 16) * DIMQ + k, DIMQ);
#pragma unroll
        for (int j = 0; j < 2; j++)
            bfr[j] = load_b_frag(bbase + (size_t)(j * 16) * DIMQ + k, DIMQ);
#pragma unroll
        for (int i = 0; i < 4; i++)
#pragma unroll
            for (int j = 0; j < 2; j++)
                acc[i][j] = wmma_bf16(af[i], bfr[j], acc[i][j]);
    }

    int lane = threadIdx.x & 31;
    int colg = lane & 15, hi = lane >> 4;
    int sec = e0 / DIMQ;              // 32-wide tile never straddles q/k/v section
    int h = (e0 % DIMQ) / HDQ;        // nor a head (64 % 32 == 0)

#pragma unroll
    for (int cj = 0; cj < 2; cj++) {
        int dl = ((e0 + cj * 16) % HDQ) + colg;   // dim within head, 0..63
#pragma unroll
        for (int ci = 0; ci < 4; ci++) {
#pragma unroll
            for (int r = 0; r < 8; r++) {
                int t = m0 + ci * 16 + r + 8 * hi;
                float val = acc[ci][cj][r];
                float partner = __shfl_xor(val, 1, 32);
                if (t < NTOK) {
                    int b = t / NQ, n = t - b * NQ;
                    if (sec < 2) {
                        if (n > 0) {
                            float c = fcos[(size_t)(n - 1) * (HDQ / 2) + (dl >> 1)];
                            float s = fsin[(size_t)(n - 1) * (HDQ / 2) + (dl >> 1)];
                            val = (dl & 1) ? (partner * s + val * c) : (val * c - partner * s);
                        }
                        if (sec == 0) val *= 0.125f;  // 1/sqrt(64)
                        __bf16* dst = (sec == 0) ? qbuf : kbuf;
                        dst[((size_t)(b * HQ + h) * NP + n) * HDQ + dl] = f2bf(val);
                    } else {
                        vT[((size_t)(b * HQ + h) * HDQ + dl) * NP + n] = f2bf(val);
                    }
                }
            }
        }
    }
}

// ---------------- stage 2: flash attention per (b,h, 16-row q tile) ------------
__global__ __launch_bounds__(32) void attn_kernel(
    const __bf16* __restrict__ qbuf, const __bf16* __restrict__ kbuf,
    const __bf16* __restrict__ vT, __bf16* __restrict__ aout)
{
    __shared__ __align__(32) __bf16 pbuf[16 * 32];

    int blk = blockIdx.x;
    const int QT = NP / 16;              // 66 query tiles
    int nt = blk % QT;
    int bh = blk / QT;
    int n0 = nt * 16;
    int lane = threadIdx.x & 31;
    int colg = lane & 15, hi = lane >> 4;

    const __bf16* qb = qbuf + ((size_t)bh * NP + n0) * HDQ;
    const __bf16* kb = kbuf + (size_t)bh * NP * HDQ;
    const __bf16* vb = vT + (size_t)bh * HDQ * NP;

    v16bf qf0 = load_a_frag(qb, HDQ);
    v16bf qf1 = load_a_frag(qb + 32, HDQ);

    const v8f vzero = {0, 0, 0, 0, 0, 0, 0, 0};
    v8f o0 = vzero, o1 = vzero, o2 = vzero, o3 = vzero;
    float m[8], l[8];
#pragma unroll
    for (int r = 0; r < 8; r++) { m[r] = -3.0e38f; l[r] = 0.f; }

    for (int j0 = 0; j0 < NP; j0 += 32) {
        v8f s0 = vzero, s1 = vzero;
        s0 = wmma_bf16(qf0, load_b_frag(kb + (size_t)j0 * HDQ, HDQ), s0);
        s0 = wmma_bf16(qf1, load_b_frag(kb + (size_t)j0 * HDQ + 32, HDQ), s0);
        s1 = wmma_bf16(qf0, load_b_frag(kb + (size_t)(j0 + 16) * HDQ, HDQ), s1);
        s1 = wmma_bf16(qf1, load_b_frag(kb + (size_t)(j0 + 16) * HDQ + 32, HDQ), s1);

        bool mask0 = (j0 + colg) >= NQ;
        bool mask1 = (j0 + 16 + colg) >= NQ;
        float p0[8], p1[8], alpha[8];
#pragma unroll
        for (int r = 0; r < 8; r++) {
            float a = mask0 ? -3.0e38f : s0[r];
            float b = mask1 ? -3.0e38f : s1[r];
            float mt = hmax16(fmaxf(a, b));
            float mnew = fmaxf(m[r], mt);
            alpha[r] = __expf(m[r] - mnew);
            p0[r] = __expf(a - mnew);
            p1[r] = __expf(b - mnew);
            l[r] = l[r] * alpha[r] + hsum16(p0[r] + p1[r]);
            m[r] = mnew;
        }
#pragma unroll
        for (int r = 0; r < 8; r++) {
            o0[r] = o0[r] * alpha[r]; o1[r] = o1[r] * alpha[r];
            o2[r] = o2[r] * alpha[r]; o3[r] = o3[r] * alpha[r];
        }
        // transpose P from C layout to A layout through LDS
#pragma unroll
        for (int r = 0; r < 8; r++) {
            int row = r + 8 * hi;
            pbuf[row * 32 + colg]      = f2bf(p0[r]);
            pbuf[row * 32 + colg + 16] = f2bf(p1[r]);
        }
        __syncthreads();
        {
            const __bf16* pp = &pbuf[(size_t)colg * 32 + hi * 8];
            uint4 c0 = *reinterpret_cast<const uint4*>(pp);
            uint4 c1 = *reinterpret_cast<const uint4*>(pp + 16);
            v16bf pf = pack_frag(c0, c1);
            o0 = wmma_bf16(pf, load_b_frag(vb + j0, NP), o0);
            o1 = wmma_bf16(pf, load_b_frag(vb + 16 * NP + j0, NP), o1);
            o2 = wmma_bf16(pf, load_b_frag(vb + 32 * NP + j0, NP), o2);
            o3 = wmma_bf16(pf, load_b_frag(vb + 48 * NP + j0, NP), o3);
        }
        __syncthreads();
    }

    int b = bh / HQ, h = bh % HQ;
#pragma unroll
    for (int r = 0; r < 8; r++) {
        int n = n0 + r + 8 * hi;
        if (n < NQ) {
            float inv = 1.0f / l[r];
            size_t base = ((size_t)(b * NQ + n)) * DIMQ + h * HDQ;
            aout[base + 0 + colg]  = f2bf(o0[r] * inv);
            aout[base + 16 + colg] = f2bf(o1[r] * inv);
            aout[base + 32 + colg] = f2bf(o2[r] * inv);
            aout[base + 48 + colg] = f2bf(o3[r] * inv);
        }
    }
}

// ---------------- stage 3: projection GEMM (64x32 per wave) + bias -------------
__global__ __launch_bounds__(32) void proj_kernel(
    const __bf16* __restrict__ abf, const __bf16* __restrict__ wbf,
    const float* __restrict__ bias, float* __restrict__ out)
{
    const int ET = DIMQ / 32;  // 24
    int blk = blockIdx.x;
    int mt = blk / ET, et = blk % ET;
    int m0 = mt * 64, e0 = et * 32;
    const __bf16* abase = abf + (size_t)m0 * DIMQ;
    const __bf16* bbase = wbf + (size_t)e0 * DIMQ;

    const v8f vzero = {0, 0, 0, 0, 0, 0, 0, 0};
    v8f acc[4][2];
#pragma unroll
    for (int i = 0; i < 4; i++)
#pragma unroll
        for (int j = 0; j < 2; j++) acc[i][j] = vzero;

    for (int k = 0; k < DIMQ; k += 32) {
        v16bf af[4], bfr[2];
#pragma unroll
        for (int i = 0; i < 4; i++)
            af[i] = load_a_frag(abase + (size_t)(i * 16) * DIMQ + k, DIMQ);
#pragma unroll
        for (int j = 0; j < 2; j++)
            bfr[j] = load_b_frag(bbase + (size_t)(j * 16) * DIMQ + k, DIMQ);
#pragma unroll
        for (int i = 0; i < 4; i++)
#pragma unroll
            for (int j = 0; j < 2; j++)
                acc[i][j] = wmma_bf16(af[i], bfr[j], acc[i][j]);
    }

    int lane = threadIdx.x & 31;
    int colg = lane & 15, hi = lane >> 4;
#pragma unroll
    for (int cj = 0; cj < 2; cj++) {
        float bv = bias[e0 + cj * 16 + colg];
#pragma unroll
        for (int ci = 0; ci < 4; ci++) {
#pragma unroll
            for (int r = 0; r < 8; r++) {
                int t = m0 + ci * 16 + r + 8 * hi;
                if (t < NTOK)
                    out[(size_t)t * DIMQ + e0 + cj * 16 + colg] = acc[ci][cj][r] + bv;
            }
        }
    }
}

extern "C" void kernel_launch(void* const* d_in, const int* in_sizes, int n_in,
                              void* d_out, int out_size, void* d_ws, size_t ws_size,
                              hipStream_t stream) {
    (void)in_sizes; (void)n_in; (void)out_size; (void)ws_size;
    const float* x     = (const float*)d_in[0];
    const float* fcos  = (const float*)d_in[1];
    const float* fsin  = (const float*)d_in[2];
    const float* wqkv  = (const float*)d_in[3];
    const float* wproj = (const float*)d_in[4];
    const float* bproj = (const float*)d_in[5];
    float* out = (float*)d_out;

    char* ws = (char*)d_ws;
    size_t off = 0;
    auto alloc = [&](size_t elems) -> char* {
        char* p = ws + off;
        off += (elems * 2 + 255) & ~(size_t)255;
        return p;
    };
    __bf16* xbf     = (__bf16*)alloc((size_t)NTOK * DIMQ);
    __bf16* wqkvbf  = (__bf16*)alloc((size_t)3 * DIMQ * DIMQ);
    __bf16* wprojbf = (__bf16*)alloc((size_t)DIMQ * DIMQ);
    __bf16* qbuf    = (__bf16*)alloc((size_t)BQ * HQ * NP * HDQ);
    __bf16* kbuf    = (__bf16*)alloc((size_t)BQ * HQ * NP * HDQ);
    __bf16* vT      = (__bf16*)alloc((size_t)BQ * HQ * HDQ * NP);
    // pad abuf by 64 rows: tail M-tile of proj GEMM reads past row 16399
    __bf16* abuf    = (__bf16*)alloc((size_t)(NTOK + 64) * DIMQ);

    // zero q/k/vT (pads must be benign; buffers are contiguous in ws)
    long long zwords = ((char*)abuf - (char*)qbuf) / 4;
    zero_kernel<<<2048, 256, 0, stream>>>((unsigned*)qbuf, zwords);

    cvt_bf16_kernel<<<2048, 256, 0, stream>>>(x, xbf, NTOK * DIMQ);
    cvt_bf16_kernel<<<1024, 256, 0, stream>>>(wqkv, wqkvbf, 3 * DIMQ * DIMQ);
    cvt_bf16_kernel<<<512, 256, 0, stream>>>(wproj, wprojbf, DIMQ * DIMQ);

    const int MT = (NTOK + 63) / 64;  // 257
    // QKV + RoPE: 257 token tiles x 72 output tiles, one wave each
    qkv_rope_kernel<<<dim3(MT * ((3 * DIMQ) / 32)), 32, 0, stream>>>(
        xbf, wqkvbf, fcos, fsin, qbuf, kbuf, vT);

    // attention: (b,h) x 66 query tiles of 16
    attn_kernel<<<dim3(BQ * HQ * (NP / 16)), 32, 0, stream>>>(qbuf, kbuf, vT, abuf);

    // projection: 257 token tiles x 24 output tiles
    proj_kernel<<<dim3(MT * (DIMQ / 32)), 32, 0, stream>>>(
        abuf, wprojbf, bproj, out);
}
